// TrueQLACI_69217692942943
// MI455X (gfx1250) — compile-verified, matrix operand order checked by reading, not run on hardware
//
#include <hip/hip_runtime.h>
#include <hip/hip_bf16.h>

typedef __attribute__((ext_vector_type(16))) _Float16 v16h;
typedef __attribute__((ext_vector_type(8)))  float    v8f;

#define B_TOTAL 262144
#define T_TOK   17
#define D_IN    8
#define D_H     32
#define D_Q     4
#define AMP2    0.0625f   // 0.25^2

// ---------------------------------------------------------------------------
// Setup: fold the two linear layers into 4x8 effective projections.
//  ws[0..31]  = WqT_eff[d][i]  (d=0..7, i=0..3)   Wq_eff = W_q @ W_emb
//  ws[32..35] = bq_eff[i]                          = W_q @ b_emb + b_q
//  ws[36..67] = WkT_eff[d][i]
//  ws[68..71] = bk_eff[i]
// ---------------------------------------------------------------------------
__global__ void fold_weights_kernel(const float* __restrict__ W_emb,
                                    const float* __restrict__ b_emb,
                                    const float* __restrict__ W_q,
                                    const float* __restrict__ b_q,
                                    const float* __restrict__ W_k,
                                    const float* __restrict__ b_k,
                                    float* __restrict__ ws) {
    int tid = threadIdx.x;            // single block of 128 threads
    if (tid < 32) {
        int d = tid >> 2, i = tid & 3;
        float acc = 0.f;
        for (int h = 0; h < D_H; ++h) acc += W_q[i * D_H + h] * W_emb[h * D_IN + d];
        ws[tid] = acc;
    } else if (tid < 36) {
        int i = tid - 32;
        float acc = b_q[i];
        for (int h = 0; h < D_H; ++h) acc += W_q[i * D_H + h] * b_emb[h];
        ws[tid] = acc;
    } else if (tid < 68) {
        int t = tid - 36;
        int d = t >> 2, i = t & 3;
        float acc = 0.f;
        for (int h = 0; h < D_H; ++h) acc += W_k[i * D_H + h] * W_emb[h * D_IN + d];
        ws[tid] = acc;
    } else if (tid < 72) {
        int i = tid - 68;
        float acc = b_k[i];
        for (int h = 0; h < D_H; ++h) acc += W_k[i * D_H + h] * b_emb[h];
        ws[tid] = acc;
    }
}

// ---------------------------------------------------------------------------
// Main kernel: 16 batches per wave, 8 waves (128 batches) per 256-thread block.
// Lane l handles batch m = l&15, past-token range n0 = (l>>4)*8 .. +7.
// s = theta @ M^T done with one v_wmma_f32_16x16x32_f16 per tile.
// ---------------------------------------------------------------------------
__global__ void __launch_bounds__(256)
needle_qlaci_kernel(const float* __restrict__ tokens,
                    const float* __restrict__ fw,     // 72 folded floats
                    const float* __restrict__ W_out,  // 32 floats
                    const float* __restrict__ b_out,  // 32 floats
                    float* __restrict__ out) {
    const int  lane = threadIdx.x & 31;
    const int  wave = threadIdx.x >> 5;
    const long b0   = (long)blockIdx.x * 128 + (long)wave * 16;
    const int  m    = lane & 15;          // batch within the 16-batch tile
    const int  n0   = (lane >> 4) * 8;    // this lane's past-token range

    const float* tb = tokens + (b0 + m) * (long)(T_TOK * D_IN);

    // needle token (t = 16)
    float t16[8];
#pragma unroll
    for (int d = 0; d < 8; ++d) t16[d] = tb[16 * D_IN + d];

    // q = Wq_eff @ t16 + bq_eff
    float q0 = fw[32], q1 = fw[33], q2 = fw[34], q3 = fw[35];
#pragma unroll
    for (int d = 0; d < 8; ++d) {
        float t = t16[d];
        q0 = fmaf(fw[d * 4 + 0], t, q0);
        q1 = fmaf(fw[d * 4 + 1], t, q1);
        q2 = fmaf(fw[d * 4 + 2], t, q2);
        q3 = fmaf(fw[d * 4 + 3], t, q3);
    }

    // r = Wk_eff^T q  (8 floats), c = bk_eff . q  -> theta[n] = tok_n . r + c
    float r[8];
#pragma unroll
    for (int d = 0; d < 8; ++d)
        r[d] = fw[36 + d * 4 + 0] * q0 + fw[36 + d * 4 + 1] * q1 +
               fw[36 + d * 4 + 2] * q2 + fw[36 + d * 4 + 3] * q3;
    float c = fw[68] * q0 + fw[69] * q1 + fw[70] * q2 + fw[71] * q3;

    // A fragment: theta[m, n0..n0+7] in f16, K=16..31 zero-padded
    v16h afrag;
#pragma unroll
    for (int j = 0; j < 8; ++j) {
        const float* tn = tb + (n0 + j) * D_IN;
        float th = c;
#pragma unroll
        for (int d = 0; d < 8; ++d) th = fmaf(tn[d], r[d], th);
        afrag[j] = (_Float16)th;
    }
#pragma unroll
    for (int j = 8; j < 16; ++j) afrag[j] = (_Float16)0.0f;

    // B fragment: signs M[a, n] with a = lane&15, n = n0 + j
    v16h bfrag;
#pragma unroll
    for (int j = 0; j < 8; ++j) {
        int par = __popc(m & (n0 + j)) & 1;
        bfrag[j] = par ? (_Float16)(-1.0f) : (_Float16)(1.0f);
    }
#pragma unroll
    for (int j = 8; j < 16; ++j) bfrag[j] = (_Float16)0.0f;

    // s tile (16 batches x 16 addresses) = theta @ M^T
    v8f cfrag = {};
    v8f s = __builtin_amdgcn_wmma_f32_16x16x32_f16(
        /*neg_a=*/false, afrag, /*neg_b=*/false, bfrag,
        /*c_mod=*/(short)0, cfrag, /*reuse_a=*/false, /*reuse_b=*/false);

    // cos^2(s/2) + sin^2(s/2), reduce over addresses (16 lanes per half-wave)
    float ev[8];
#pragma unroll
    for (int v = 0; v < 8; ++v) {
        float h  = 0.5f * s[v];
        float sn = __sinf(h);
        float cs = __cosf(h);
        ev[v] = fmaf(sn, sn, cs * cs);
    }
#pragma unroll
    for (int mask = 1; mask <= 8; mask <<= 1) {
#pragma unroll
        for (int v = 0; v < 8; ++v) ev[v] += __shfl_xor(ev[v], mask, 32);
    }
    // ev[v] = sum over a for batch (v + 8*(lane>=16)), uniform within each half

    // output head: out[b][h] = expval[b] * W_out[h] + b_out[h], h = lane
    const float wout = W_out[lane];
    const float bout = b_out[lane];
    float* orow = out + b0 * D_H;
#pragma unroll
    for (int b = 0; b < 16; ++b) {
        float e = AMP2 * __shfl(ev[b & 7], (b >> 3) * 16, 32);
        orow[b * D_H + lane] = fmaf(e, wout, bout);
    }
}

// ---------------------------------------------------------------------------
extern "C" void kernel_launch(void* const* d_in, const int* in_sizes, int n_in,
                              void* d_out, int out_size, void* d_ws, size_t ws_size,
                              hipStream_t stream) {
    const float* tokens = (const float*)d_in[0];
    const float* W_emb  = (const float*)d_in[1];
    const float* b_emb  = (const float*)d_in[2];
    const float* W_q    = (const float*)d_in[3];
    const float* b_q    = (const float*)d_in[4];
    const float* W_k    = (const float*)d_in[5];
    const float* b_k    = (const float*)d_in[6];
    const float* W_out  = (const float*)d_in[7];
    const float* b_out  = (const float*)d_in[8];
    float*       out    = (float*)d_out;
    float*       fw     = (float*)d_ws;   // 72 floats of folded weights

    fold_weights_kernel<<<1, 128, 0, stream>>>(W_emb, b_emb, W_q, b_q, W_k, b_k, fw);
    needle_qlaci_kernel<<<B_TOTAL / 128, 256, 0, stream>>>(tokens, fw, W_out, b_out, out);
}
